// RNN_42219528520197
// MI455X (gfx1250) — matrix operand order repaired and checked by reading, not compiled, not run on hardware
//
#include <hip/hip_runtime.h>

typedef __attribute__((ext_vector_type(2)))  float  v2f;
typedef __attribute__((ext_vector_type(8)))  float  v8f;
typedef __attribute__((ext_vector_type(8)))  __bf16 v8bf;
typedef __attribute__((ext_vector_type(16))) __bf16 v16bf;

#define B_ 16
#define S_ 4096
#define D_ 256
#define H_ 256
#define KSTEPS 64    // 256 / 4   (fp32 WMMA K-steps, xproj)
#define KK 8         // 256 / 32  (bf16 WMMA K-steps, scan)
#define NTILES 16    // 256 / 16
#define BPAD 264     // bf16 LDS row stride: 528B -> 4-dword bank rotation

// ---------------------------------------------------------------------------
// Kernel 1: x_proj[s] = x[:, s, :] @ Wx + b  in exact fp32 (HBM-bound, so
// full precision is free). Output stored in raw WMMA C-fragment layout:
// xp[((s*16 + n)*32 + lane)*8 + v], one v8f per lane per 16x16 tile.
// ---------------------------------------------------------------------------
__global__ __launch_bounds__(512)
void rnn_xproj_kernel(const float* __restrict__ x,    // [B, S, D]
                      const float* __restrict__ Wx,   // [D, H]
                      const float* __restrict__ bias, // [H]
                      float* __restrict__ xp)         // [S, 16, 32, 8] frags
{
    const int wave = threadIdx.x >> 5;   // n-tile 0..15
    const int lane = threadIdx.x & 31;
    const int lg   = lane >> 4;          // 0 => K+{0,1}, 1 => K+{2,3}
    const int lm   = lane & 15;
    const int s    = blockIdx.x;
    const int h0   = wave * 16;

    const float* __restrict__ arow =
        x + (size_t)lm * (S_ * D_) + (size_t)s * D_;

    v8f acc0 = {}, acc1 = {};            // two chains for load/WMMA overlap
#pragma unroll 8
    for (int k = 0; k < KSTEPS; k += 2) {
        const int d0 = 4 * k + 2 * lg;
        const int d1 = 4 * (k + 1) + 2 * lg;
        v2f a0 = *(const v2f*)(arow + d0);
        v2f a1 = *(const v2f*)(arow + d1);
        v2f b0, b1;
        b0.x = Wx[(size_t)d0 * H_ + h0 + lm];
        b0.y = Wx[(size_t)(d0 + 1) * H_ + h0 + lm];
        b1.x = Wx[(size_t)d1 * H_ + h0 + lm];
        b1.y = Wx[(size_t)(d1 + 1) * H_ + h0 + lm];
        acc0 = __builtin_amdgcn_wmma_f32_16x16x4_f32(
            false, a0, false, b0, (short)0, acc0, false, false);
        acc1 = __builtin_amdgcn_wmma_f32_16x16x4_f32(
            false, a1, false, b1, (short)0, acc1, false, false);
    }

    const float bb = bias[h0 + lm];      // N = h0 + lane%16 for every C VGPR
    v8f res;
#pragma unroll
    for (int v = 0; v < 8; ++v) res[v] = acc0[v] + acc1[v] + bb;

    *(v8f*)(xp + ((size_t)(s * NTILES + wave) * 32 + lane) * 8) = res;
}

// ---------------------------------------------------------------------------
// Kernel 2: sequential scan  h = tanh(xp_t + h @ Wh).  1 workgroup, 16 waves,
// wave j owns output columns [16j, 16j+16).
// h@Wh runs on V_WMMA_F32_16X16X32_BF16 with a hi/lo split-precision scheme:
//   h = h_hi + h_lo,  Wh = W_hi + W_lo  (bf16 pairs)
//   h@Wh ~= h_hi*W_hi + h_hi*W_lo + h_lo*W_hi   (3 independent f32 chains)
// => 24 WMMAs per wave per step (vs 64 on the f32 K=4 path), ~2^-16 rel err.
// Wh panels live in 128 VGPRs of v16bf; h ping-pongs in padded bf16 LDS.
// ---------------------------------------------------------------------------
__global__ __launch_bounds__(512, 1)
void rnn_scan_kernel(const float* __restrict__ xp,     // fragment layout
                     const float* __restrict__ Wh,     // [H, H] fp32
                     const float* __restrict__ h0in,   // [B, H] state0
                     float* __restrict__ out)          // [B, S, H]
{
    __shared__ __bf16 hhi[2][16 * BPAD];   // 2 x 8.25 KB
    __shared__ __bf16 hlo[2][16 * BPAD];   // 2 x 8.25 KB

    const int wave = threadIdx.x >> 5;
    const int lane = threadIdx.x & 31;
    const int lg   = lane >> 4;
    const int lm   = lane & 15;
    const int h0   = wave * 16;

    // --- Build this wave's Wh panel as bf16 hi/lo B-fragments (registers).
    // B panel, contiguous k order chosen by this code: lane holds column N = h0+lm; element e of the
    // v16bf is K = 32*kk + 16*lg + e (2 values per VGPR).
    v16bf bhi[KK], blo[KK];
#pragma unroll
    for (int kk = 0; kk < KK; ++kk) {
        v16bf bh, bl;
#pragma unroll
        for (int e = 0; e < 16; ++e) {
            const int K = 32 * kk + 16 * lg + e;
            const float w = Wh[(size_t)K * H_ + h0 + lm];
            const __bf16 hi = (__bf16)w;
            bh[e] = hi;
            bl[e] = (__bf16)(w - (float)hi);
        }
        bhi[kk] = bh;
        blo[kk] = bl;
    }

    // --- Init h buffer 0 from state0 (split into hi/lo) ---
    for (int i = threadIdx.x; i < B_ * H_; i += 512) {
        const int r = i >> 8, c = i & 255;
        const float w = h0in[i];
        const __bf16 hi = (__bf16)w;
        hhi[0][r * BPAD + c] = hi;
        hlo[0][r * BPAD + c] = (__bf16)(w - (float)hi);
    }
    __syncthreads();

    int p = 0;
    for (int s = 0; s < S_; ++s) {
        // xp fragment: one coalesced 32B load per lane (L2-resident)
        const v8f xpf = *(const v8f*)(xp +
            ((size_t)(s * NTILES + wave) * 32 + lane) * 8);

        const __bf16* __restrict__ Hh = &hhi[p][0];
        const __bf16* __restrict__ Hl = &hlo[p][0];
        v8f c0 = {}, c1 = {}, c2 = {};   // hi*hi, hi*lo, lo*hi chains

#pragma unroll
        for (int kk = 0; kk < KK; ++kk) {
            // A fragment, contiguous k order chosen by this code: lane = row M=lm; element e is
            // K = 32kk + 16*lg + e (matches the B panel order)
            const int base = lm * BPAD + 32 * kk + 16 * lg;
            v8bf ah0 = *(const v8bf*)(Hh + base);
            v8bf ah1 = *(const v8bf*)(Hh + base + 8);
            v8bf al0 = *(const v8bf*)(Hl + base);
            v8bf al1 = *(const v8bf*)(Hl + base + 8);
            v16bf ahi = __builtin_shufflevector(ah0, ah1,
                0,1,2,3,4,5,6,7,8,9,10,11,12,13,14,15);
            v16bf alo = __builtin_shufflevector(al0, al1,
                0,1,2,3,4,5,6,7,8,9,10,11,12,13,14,15);
            c0 = __builtin_amdgcn_wmma_f32_16x16x32_bf16(
                false, ahi, false, bhi[kk], (short)0, c0, false, false);
            c1 = __builtin_amdgcn_wmma_f32_16x16x32_bf16(
                false, ahi, false, blo[kk], (short)0, c1, false, false);
            c2 = __builtin_amdgcn_wmma_f32_16x16x32_bf16(
                false, alo, false, bhi[kk], (short)0, c2, false, false);
        }

        __bf16* __restrict__ Nh = &hhi[p ^ 1][0];
        __bf16* __restrict__ Nl = &hlo[p ^ 1][0];
#pragma unroll
        for (int v = 0; v < 8; ++v) {
            float t = xpf[v] + ((c0[v] + c1[v]) + c2[v]);
            t = tanhf(t);
            const int brow = v + 8 * lg;           // C layout: M = v + 8*lg
            const __bf16 hi = (__bf16)t;
            Nh[brow * BPAD + h0 + lm] = hi;        // next-step h (hi)
            Nl[brow * BPAD + h0 + lm] = (__bf16)(t - (float)hi); // (lo)
            __builtin_nontemporal_store(            // streamed fp32 output
                t, out + (size_t)brow * (S_ * H_) + (size_t)s * H_ + h0 + lm);
        }
        __syncthreads();
        p ^= 1;
    }
}

// ---------------------------------------------------------------------------
extern "C" void kernel_launch(void* const* d_in, const int* in_sizes, int n_in,
                              void* d_out, int out_size, void* d_ws, size_t ws_size,
                              hipStream_t stream) {
    (void)in_sizes; (void)n_in; (void)out_size; (void)ws_size;
    const float* x      = (const float*)d_in[0];   // [B,S,D]
    const float* state0 = (const float*)d_in[1];   // [B,H]
    const float* Wx     = (const float*)d_in[2];   // [D,H]
    const float* Wh     = (const float*)d_in[3];   // [H,H]
    const float* bias   = (const float*)d_in[4];   // [H]
    float* out = (float*)d_out;                    // [B,S,H]
    float* xpf = (float*)d_ws;                     // 64 MB fragment workspace

    rnn_xproj_kernel<<<S_, 512, 0, stream>>>(x, Wx, bias, xpf);
    rnn_scan_kernel<<<1, 512, 0, stream>>>(xpf, Wh, state0, out);
}
